// MultiHeadAttention_61753039781966
// MI455X (gfx1250) — compile-verified, matrix operand order checked by reading, not run on hardware
//
#include <hip/hip_runtime.h>
#include <hip/hip_bf16.h>

typedef __bf16 bf16_t;
typedef __attribute__((ext_vector_type(8)))  __bf16 v8bf;
typedef __attribute__((ext_vector_type(16))) __bf16 v16bf;
typedef __attribute__((ext_vector_type(8)))  float  v8f;

__device__ __forceinline__ v8bf ldg8(const bf16_t* p) { return *(const v8bf*)p; }

__device__ __forceinline__ v16bf cat16(v8bf lo, v8bf hi) {
  return __builtin_shufflevector(lo, hi, 0,1,2,3,4,5,6,7,8,9,10,11,12,13,14,15);
}

__device__ __forceinline__ v8f wmma_bf16(v16bf a, v16bf b, v8f c) {
  // D = A(16x32) * B(32x16) + C, fp32 accumulate
  return __builtin_amdgcn_wmma_f32_16x16x32_bf16(false, a, false, b, (short)0, c, false, false);
}

// A-fragment (16x32 bf16) built from fp32 row: elements 0..7 = k+half8..,
// 8..15 = k+16+half8..  (matches ISA 16-bit A packing)
__device__ __forceinline__ v16bf load_a_f32(const float* row, int k, int half8) {
  float4 a0 = *(const float4*)(row + k + half8);
  float4 a1 = *(const float4*)(row + k + half8 + 4);
  float4 a2 = *(const float4*)(row + k + 16 + half8);
  float4 a3 = *(const float4*)(row + k + 16 + half8 + 4);
  v16bf a;
  a[0]=(bf16_t)a0.x; a[1]=(bf16_t)a0.y; a[2]=(bf16_t)a0.z; a[3]=(bf16_t)a0.w;
  a[4]=(bf16_t)a1.x; a[5]=(bf16_t)a1.y; a[6]=(bf16_t)a1.z; a[7]=(bf16_t)a1.w;
  a[8]=(bf16_t)a2.x; a[9]=(bf16_t)a2.y; a[10]=(bf16_t)a2.z; a[11]=(bf16_t)a2.w;
  a[12]=(bf16_t)a3.x; a[13]=(bf16_t)a3.y; a[14]=(bf16_t)a3.z; a[15]=(bf16_t)a3.w;
  return a;
}

// ---------------------------------------------------------------------------
// Kernel 1: transpose + convert weights to bf16, [K,N] -> [N,K]
// ---------------------------------------------------------------------------
__global__ void cvtw_kernel(const float* __restrict__ Wqkv, const float* __restrict__ Wout,
                            bf16_t* __restrict__ Wqkv_t, bf16_t* __restrict__ Wout_t) {
  const int D = 1024, N3 = 3072;
  int idx = blockIdx.x * blockDim.x + threadIdx.x;
  const int total1 = D * N3;                       // 3,145,728
  if (idx < total1) {
    int n = idx / D, k = idx - n * D;
    Wqkv_t[idx] = (bf16_t)Wqkv[(size_t)k * N3 + n];
  } else {
    int i2 = idx - total1;                         // < D*D
    int n = i2 / D, k = i2 - n * D;
    Wout_t[i2] = (bf16_t)Wout[(size_t)k * D + n];
  }
}

// ---------------------------------------------------------------------------
// Kernel 2: qkv = x @ W_qkv + b ; scatter into Q[B,H,S,64], K[B,H,S,64],
// V transposed [B,H,64,S] (all bf16). One wave -> 32(M) x 64(N) tile:
// 8 WMMAs per K=32 step sharing 4 B-fragments between 2 A-fragments.
// ---------------------------------------------------------------------------
__global__ void qkv_kernel(const float* __restrict__ x, const bf16_t* __restrict__ Wt,
                           const float* __restrict__ bias,
                           bf16_t* __restrict__ Qb, bf16_t* __restrict__ Kb,
                           bf16_t* __restrict__ Vt) {
  const int S = 2048, D = 1024, H = 16, Hd = 64, NT = 48; // 3072/64
  int wave   = blockIdx.x * (blockDim.x >> 5) + (threadIdx.x >> 5);
  int mt     = wave / NT, nt = wave - mt * NT;
  int mbase  = mt * 32;        // flattened token index (b*S+s)
  int nbase  = nt * 64;
  int lane   = threadIdx.x & 31;
  int lane16 = lane & 15;
  int half8  = (lane & 16) ? 8 : 0;

  v8f acc[2][4] = {{v8f{}, v8f{}, v8f{}, v8f{}}, {v8f{}, v8f{}, v8f{}, v8f{}}};
  const float* xrow0 = x + (size_t)(mbase + lane16) * D;
  const float* xrow1 = xrow0 + (size_t)16 * D;

  for (int k = 0; k < D; k += 32) {
    v16bf a0 = load_a_f32(xrow0, k, half8);
    v16bf a1 = load_a_f32(xrow1, k, half8);
#pragma unroll
    for (int j = 0; j < 4; ++j) {
      const bf16_t* wrow = Wt + (size_t)(nbase + j * 16 + lane16) * D + k + half8;
      v16bf b = cat16(ldg8(wrow), ldg8(wrow + 16));
      acc[0][j] = wmma_bf16(a0, b, acc[0][j]);
      acc[1][j] = wmma_bf16(a1, b, acc[1][j]);
    }
  }

#pragma unroll
  for (int j = 0; j < 4; ++j) {
    int n = nbase + j * 16 + lane16;          // column in [0,3072)
    float bv = bias[n];
    int which = n >> 10;                      // 0=q 1=k 2=v (wave-uniform per j)
    int d = n & 1023;
    int h = d >> 6, hd = d & 63;
#pragma unroll
    for (int mi = 0; mi < 2; ++mi) {
#pragma unroll
      for (int r = 0; r < 8; ++r) {
        int m = mbase + mi * 16 + r + half8;  // token
        int bb = m >> 11, s = m & 2047;
        bf16_t v = (bf16_t)(acc[mi][j][r] + bv);
        size_t bh = (size_t)bb * H + h;
        if (which == 0)      Qb[(bh * S + s) * Hd + hd] = v;
        else if (which == 1) Kb[(bh * S + s) * Hd + hd] = v;
        else                 Vt[(bh * Hd + hd) * S + s] = v;
      }
    }
  }
}

// ---------------------------------------------------------------------------
// Kernel 3: causal flash attention. One wave per (b,h, 32-query block):
// two 16-query tiles share every K/V fragment. Computes S^T = K*Q^T so
// softmax rows are per-lane; C-layout of exp(S^T) is exactly the B-operand
// layout of P^T for ctx^T = V^T * P^T. No LDS, no cross-lane transposes.
// ---------------------------------------------------------------------------
__global__ void attn_kernel(const bf16_t* __restrict__ Qb, const bf16_t* __restrict__ Kb,
                            const bf16_t* __restrict__ Vt, bf16_t* __restrict__ ctx) {
  const int S = 2048, H = 16, Hd = 64, D = 1024, QB = S / 32; // 64 blocks of 32 queries
  int wave   = blockIdx.x * (blockDim.x >> 5) + (threadIdx.x >> 5);
  int bh     = wave / QB;
  int qb     = wave - bh * QB;
  int b      = bh >> 4, h = bh & 15;
  int lane   = threadIdx.x & 31;
  int lane16 = lane & 15;
  int half8  = (lane & 16) ? 8 : 0;
  int qbase  = qb * 32;

  const bf16_t* Qh = Qb + (size_t)bh * S * Hd;
  const bf16_t* Kh = Kb + (size_t)bh * S * Hd;
  const bf16_t* Vh = Vt + (size_t)bh * Hd * S;

  // Q^T B-fragments for the two query tiles
  const bf16_t* qrowA = Qh + (size_t)(qbase + lane16) * Hd + half8;
  const bf16_t* qrowB = qrowA + (size_t)16 * Hd;
  v16bf qfA0 = cat16(ldg8(qrowA),      ldg8(qrowA + 16));  // feats 0..31
  v16bf qfA1 = cat16(ldg8(qrowA + 32), ldg8(qrowA + 48));  // feats 32..63
  v16bf qfB0 = cat16(ldg8(qrowB),      ldg8(qrowB + 16));
  v16bf qfB1 = cat16(ldg8(qrowB + 32), ldg8(qrowB + 48));

  const float scale = 0.125f;                 // 1/sqrt(64)
  float mA = -3.0e38f, lA = 0.0f, mB = -3.0e38f, lB = 0.0f;
  v8f accA[4] = {v8f{}, v8f{}, v8f{}, v8f{}};
  v8f accB[4] = {v8f{}, v8f{}, v8f{}, v8f{}};
  int qA = qbase + lane16, qB = qbase + 16 + lane16;

  int nch = qb + 1;                           // 32-key chunks covering keys <= qbase+31
  for (int c = 0; c < nch; ++c) {
    int kb = c * 32;
    float pA[2][8], pB[2][8];
    float cmA = -3.0e38f, cmB = -3.0e38f;
#pragma unroll
    for (int kt = 0; kt < 2; ++kt) {
      const bf16_t* krow = Kh + (size_t)(kb + kt * 16 + lane16) * Hd + half8;
      v16bf kf0 = cat16(ldg8(krow),      ldg8(krow + 16));
      v16bf kf1 = cat16(ldg8(krow + 32), ldg8(krow + 48));
      v8f stA = {}, stB = {};
      stA = wmma_bf16(kf0, qfA0, stA);
      stA = wmma_bf16(kf1, qfA1, stA);
      stB = wmma_bf16(kf0, qfB0, stB);
      stB = wmma_bf16(kf1, qfB1, stB);
#pragma unroll
      for (int r = 0; r < 8; ++r) {           // key = kb + kt*16 + r + half8
        int key = kb + kt * 16 + r + half8;
        float vA = stA[r] * scale;
        float vB = stB[r] * scale;
        vA = (key <= qA) ? vA : -3.0e38f;     // causal mask
        vB = (key <= qB) ? vB : -3.0e38f;
        pA[kt][r] = vA;  cmA = fmaxf(cmA, vA);
        pB[kt][r] = vB;  cmB = fmaxf(cmB, vB);
      }
    }
    cmA = fmaxf(cmA, __shfl_xor(cmA, 16, 32));
    cmB = fmaxf(cmB, __shfl_xor(cmB, 16, 32));
    float mnA = fmaxf(mA, cmA), mnB = fmaxf(mB, cmB);
    float alA = __expf(mA - mnA), alB = __expf(mB - mnB);
    float rsA = 0.0f, rsB = 0.0f;
    v16bf pbA, pbB;                           // P^T B-fragments, built in place
#pragma unroll
    for (int r = 0; r < 8; ++r) {
      float a0 = __expf(pA[0][r] - mnA);
      float a1 = __expf(pA[1][r] - mnA);
      float b0 = __expf(pB[0][r] - mnB);
      float b1 = __expf(pB[1][r] - mnB);
      rsA += a0 + a1;  rsB += b0 + b1;
      pbA[r] = (bf16_t)a0;  pbA[r + 8] = (bf16_t)a1;  // keys half8+r, 16+half8+r
      pbB[r] = (bf16_t)b0;  pbB[r + 8] = (bf16_t)b1;
    }
    rsA += __shfl_xor(rsA, 16, 32);
    rsB += __shfl_xor(rsB, 16, 32);
    lA = lA * alA + rsA;  mA = mnA;
    lB = lB * alB + rsB;  mB = mnB;
#pragma unroll
    for (int f = 0; f < 4; ++f) {
#pragma unroll
      for (int r = 0; r < 8; ++r) { accA[f][r] *= alA; accB[f][r] *= alB; }
      const bf16_t* vrow = Vh + (size_t)(f * 16 + lane16) * S + kb + half8;
      v16bf vf = cat16(ldg8(vrow), ldg8(vrow + 16));    // V^T A-fragment
      accA[f] = wmma_bf16(vf, pbA, accA[f]);
      accB[f] = wmma_bf16(vf, pbB, accB[f]);
    }
  }

  float invA = 1.0f / lA, invB = 1.0f / lB;
  size_t baseA = ((size_t)b * S + qbase + lane16) * D + h * Hd;
  size_t baseB = baseA + (size_t)16 * D;
#pragma unroll
  for (int f = 0; f < 4; ++f)
#pragma unroll
    for (int r = 0; r < 8; ++r) {
      ctx[baseA + f * 16 + r + half8] = (bf16_t)(accA[f][r] * invA);
      ctx[baseB + f * 16 + r + half8] = (bf16_t)(accB[f][r] * invB);
    }
}

// ---------------------------------------------------------------------------
// Kernel 4: out = ctx @ W_out  (fp32 output). One wave -> 32x64 tile.
// ---------------------------------------------------------------------------
__global__ void out_kernel(const bf16_t* __restrict__ ctx, const bf16_t* __restrict__ Wt,
                           float* __restrict__ out) {
  const int D = 1024, NT = 16;                // 1024/64
  int wave   = blockIdx.x * (blockDim.x >> 5) + (threadIdx.x >> 5);
  int mt     = wave / NT, nt = wave - mt * NT;
  int mbase  = mt * 32, nbase = nt * 64;
  int lane   = threadIdx.x & 31;
  int lane16 = lane & 15;
  int half8  = (lane & 16) ? 8 : 0;

  v8f acc[2][4] = {{v8f{}, v8f{}, v8f{}, v8f{}}, {v8f{}, v8f{}, v8f{}, v8f{}}};
  const bf16_t* arow0 = ctx + (size_t)(mbase + lane16) * D + half8;
  const bf16_t* arow1 = arow0 + (size_t)16 * D;
  for (int k = 0; k < D; k += 32) {
    v16bf a0 = cat16(ldg8(arow0 + k), ldg8(arow0 + k + 16));
    v16bf a1 = cat16(ldg8(arow1 + k), ldg8(arow1 + k + 16));
#pragma unroll
    for (int j = 0; j < 4; ++j) {
      const bf16_t* wrow = Wt + (size_t)(nbase + j * 16 + lane16) * D + k + half8;
      v16bf b = cat16(ldg8(wrow), ldg8(wrow + 16));
      acc[0][j] = wmma_bf16(a0, b, acc[0][j]);
      acc[1][j] = wmma_bf16(a1, b, acc[1][j]);
    }
  }
#pragma unroll
  for (int mi = 0; mi < 2; ++mi)
#pragma unroll
    for (int j = 0; j < 4; ++j)
#pragma unroll
      for (int r = 0; r < 8; ++r)
        out[(size_t)(mbase + mi * 16 + r + half8) * D + nbase + j * 16 + lane16] = acc[mi][j][r];
}

// ---------------------------------------------------------------------------
extern "C" void kernel_launch(void* const* d_in, const int* in_sizes, int n_in,
                              void* d_out, int out_size, void* d_ws, size_t ws_size,
                              hipStream_t stream) {
  const float* x    = (const float*)d_in[0];  // [2,2048,1024]
  const float* Wqkv = (const float*)d_in[1];  // [1024,3072]
  const float* bqkv = (const float*)d_in[2];  // [3072]
  const float* Wout = (const float*)d_in[3];  // [1024,1024]
  float* out = (float*)d_out;                 // [2,2048,1024]

  char* ws = (char*)d_ws;
  bf16_t* Wqkv_t = (bf16_t*)ws;                               // 6 MB  [3072,1024]
  bf16_t* Wout_t = (bf16_t*)(ws + 6291456);                   // 2 MB  [1024,1024]
  bf16_t* Qb     = (bf16_t*)(ws + 6291456 + 2097152);         // 8 MB  [2,16,2048,64]
  bf16_t* Kb     = Qb + 4194304;                              // 8 MB
  bf16_t* Vt     = Kb + 4194304;                              // 8 MB  [2,16,64,2048]
  bf16_t* ctxb   = Vt + 4194304;                              // 8 MB  [2,2048,1024]

  cvtw_kernel<<<16384, 256, 0, stream>>>(Wqkv, Wout, Wqkv_t, Wout_t); // 4,194,304 elems
  qkv_kernel <<<768,   256, 0, stream>>>(x, Wqkv_t, bqkv, Qb, Kb, Vt); // 6144 waves, 32x64 tiles
  attn_kernel<<<256,   256, 0, stream>>>(Qb, Kb, Vt, ctxb);            // 2048 waves, 32q blocks
  out_kernel <<<256,   256, 0, stream>>>(ctxb, Wout_t, out);           // 2048 waves, 32x64 tiles
}
